// GroupedExperts_56066503082694
// MI455X (gfx1250) — matrix operand order repaired and checked by reading, not compiled, not run on hardware
//
#include <hip/hip_runtime.h>

// ---------------------------------------------------------------------------
// MoE grouped SwiGLU experts for MI455X (gfx1250, wave32, WMMA bf16)
// T=8192 tokens, K=2, E=8 experts, DIM=2048, HIDDEN=1408, C = T*K/E = 2048
// ---------------------------------------------------------------------------

typedef __attribute__((ext_vector_type(16))) __bf16        v16bf;
typedef __attribute__((ext_vector_type(8)))  float         v8f;
typedef __attribute__((ext_vector_type(8)))  unsigned int  v8u;

#define TOKENS   8192
#define TOPK     2
#define NEXP     8
#define DIM      2048
#define HIDDEN   1408
#define CAP      2048          // tokens per expert (balanced routing)
#define SLOTS    (TOKENS*TOPK) // 16384

#define BM 128
#define BN 64
#define BK 32
#define LDSTRIDE 40            // ushort stride per row: 32 payload + 8 pad (80B)

// ---- fp32 -> bf16 round-to-nearest-even -----------------------------------
__device__ __forceinline__ unsigned short f2bf(float f) {
    unsigned int u = __float_as_uint(f);
    u += 0x7FFFu + ((u >> 16) & 1u);
    return (unsigned short)(u >> 16);
}
__device__ __forceinline__ unsigned int pack2(float a, float b) {
    return (unsigned int)f2bf(a) | ((unsigned int)f2bf(b) << 16);
}

// ---- WMMA fragment loads from LDS (padded row-major [row][LDSTRIDE]) -------
// A-matrix 16x32 bf16: lane m=l%16; lanes0-15: K 0..7 (v0-3), 16..23 (v4-7);
//                      lanes16-31:  K 8..15,      24..31.
__device__ __forceinline__ v16bf load_fragA(const unsigned short* base, int lane) {
    int r  = lane & 15;
    int kh = (lane < 16) ? 0 : 8;
    const unsigned short* p = base + r * LDSTRIDE;
    uint4 lo = *(const uint4*)(p + kh);        // K kh..kh+7
    uint4 hi = *(const uint4*)(p + kh + 16);   // K kh+16..kh+23
    v8u u; u[0]=lo.x; u[1]=lo.y; u[2]=lo.z; u[3]=lo.w;
           u[4]=hi.x; u[5]=hi.y; u[6]=hi.z; u[7]=hi.w;
    return __builtin_bit_cast(v16bf, u);
}
// B-matrix 32x16 bf16: lane n=l%16; lanes0-15 hold K 0..7,16..23, lanes16-31 K 8..15,24..31
__device__ __forceinline__ v16bf load_fragB(const unsigned short* base, int lane) {
    int n  = lane & 15;
    int kh = (lane < 16) ? 0 : 8;
    const unsigned short* p = base + n * LDSTRIDE + kh;
    uint4 lo = *(const uint4*)(p);             // K kh..kh+7
    uint4 hi = *(const uint4*)(p + 16);        // K kh+16..kh+23
    v8u u; u[0]=lo.x; u[1]=lo.y; u[2]=lo.z; u[3]=lo.w;
           u[4]=hi.x; u[5]=hi.y; u[6]=hi.z; u[7]=hi.w;
    return __builtin_bit_cast(v16bf, u);
}

// ---------------------------------------------------------------------------
// Kernel 1: stable dispatch (counting sort by expert, reproduces stable argsort)
// ---------------------------------------------------------------------------
__global__ __launch_bounds__(256) void moe_dispatch(
    const int* __restrict__ sel, const float* __restrict__ scores,
    int* __restrict__ row_map, float* __restrict__ gate)
{
    __shared__ int hist[NEXP * 256];
    __shared__ int totals[NEXP];
    __shared__ int base[NEXP];
    const int tid = threadIdx.x;
    const int CH  = SLOTS / 256;   // 64 slots per thread, contiguous => stable

    int cnt[NEXP];
#pragma unroll
    for (int e = 0; e < NEXP; ++e) cnt[e] = 0;
    const int s0 = tid * CH;
    for (int i = 0; i < CH; ++i) cnt[sel[s0 + i] & (NEXP - 1)]++;
#pragma unroll
    for (int e = 0; e < NEXP; ++e) hist[e * 256 + tid] = cnt[e];
    __syncthreads();

    if (tid < NEXP) {                    // exclusive scan over threads, per expert
        int run = 0;
        for (int t = 0; t < 256; ++t) {
            int c = hist[tid * 256 + t];
            hist[tid * 256 + t] = run;
            run += c;
        }
        totals[tid] = run;
    }
    __syncthreads();
    if (tid == 0) {                      // exclusive scan over experts
        int acc = 0;
        for (int e = 0; e < NEXP; ++e) { base[e] = acc; acc += totals[e]; }
    }
    __syncthreads();

    int off[NEXP];
#pragma unroll
    for (int e = 0; e < NEXP; ++e) off[e] = base[e] + hist[e * 256 + tid];
    for (int i = 0; i < CH; ++i) {
        int s = s0 + i;
        int e = sel[s] & (NEXP - 1);
        int r = off[e]++;
        row_map[r] = s;                  // slot id; token = s / TOPK
        gate[r]    = scores[s];
    }
}

// ---------------------------------------------------------------------------
// Kernel 2: zero output (scatter-add target)
// ---------------------------------------------------------------------------
__global__ __launch_bounds__(256) void zero_out(float4* __restrict__ p, int n4)
{
    int i = blockIdx.x * 256 + threadIdx.x;
    if (i < n4) p[i] = make_float4(0.f, 0.f, 0.f, 0.f);
}

// ---------------------------------------------------------------------------
// Kernel 3: GEMM1 fused SwiGLU: h = silu(xg @ w1^T) * (xg @ w3^T)   (bf16 out)
// grid: (HIDDEN/BN, CAP/BM, NEXP), block 256 (8 waves, 4x2 wave grid)
// ---------------------------------------------------------------------------
__global__ __launch_bounds__(256) void moe_gemm1(
    const float* __restrict__ x,  const float* __restrict__ w1,
    const float* __restrict__ w3, const int* __restrict__ row_map,
    unsigned short* __restrict__ h)
{
    __shared__ unsigned short aS [BM * LDSTRIDE];
    __shared__ unsigned short b1S[BN * LDSTRIDE];
    __shared__ unsigned short b3S[BN * LDSTRIDE];
    __shared__ int tokS[BM];

    const int e    = blockIdx.z;
    const int m0   = blockIdx.y * BM;
    const int n0   = blockIdx.x * BN;
    const int tid  = threadIdx.x;
    const int lane = tid & 31;
    const int wv   = tid >> 5;
    const int wm   = wv & 3;         // 0..3  -> 32-row strip
    const int wn   = wv >> 2;        // 0..1  -> 32-col strip

    if (tid < BM) tokS[tid] = row_map[e * CAP + m0 + tid] >> 1;  // / TOPK
    __syncthreads();

    const float* w1e = w1 + (size_t)e * HIDDEN * DIM;
    const float* w3e = w3 + (size_t)e * HIDDEN * DIM;

    v8f acc1[2][2] = {};
    v8f acc3[2][2] = {};

    for (int k0 = 0; k0 < DIM; k0 += BK) {
        // ---- A tile: gather 128x32 fp32 -> bf16 LDS -----------------------
#pragma unroll
        for (int t = 0; t < 4; ++t) {
            int f   = tid + t * 256;         // 0..1023 float4 slots
            int row = f >> 3;
            int c4  = f & 7;
            float4 v = *(const float4*)(x + (size_t)tokS[row] * DIM + k0 + c4 * 4);
            *(uint2*)(aS + row * LDSTRIDE + c4 * 4) =
                make_uint2(pack2(v.x, v.y), pack2(v.z, v.w));
        }
        // ---- B tiles: w1 & w3, 64x32 fp32 -> bf16 LDS ---------------------
#pragma unroll
        for (int t = 0; t < 2; ++t) {
            int f   = tid + t * 256;         // 0..511
            int row = f >> 3;
            int c4  = f & 7;
            size_t off = (size_t)(n0 + row) * DIM + k0 + c4 * 4;
            float4 v1 = *(const float4*)(w1e + off);
            float4 v3 = *(const float4*)(w3e + off);
            *(uint2*)(b1S + row * LDSTRIDE + c4 * 4) =
                make_uint2(pack2(v1.x, v1.y), pack2(v1.z, v1.w));
            *(uint2*)(b3S + row * LDSTRIDE + c4 * 4) =
                make_uint2(pack2(v3.x, v3.y), pack2(v3.z, v3.w));
            if (k0 + BK < DIM) {             // gfx1250 global_prefetch next k-tile
                __builtin_prefetch(w1e + off + BK, 0, 1);
                __builtin_prefetch(w3e + off + BK, 0, 1);
            }
        }
        __syncthreads();

        v16bf a[2], b1[2], b3[2];
#pragma unroll
        for (int i = 0; i < 2; ++i)
            a[i] = load_fragA(aS + (wm * 32 + i * 16) * LDSTRIDE, lane);
#pragma unroll
        for (int j = 0; j < 2; ++j) {
            b1[j] = load_fragB(b1S + (wn * 32 + j * 16) * LDSTRIDE, lane);
            b3[j] = load_fragB(b3S + (wn * 32 + j * 16) * LDSTRIDE, lane);
        }
#pragma unroll
        for (int i = 0; i < 2; ++i)
#pragma unroll
            for (int j = 0; j < 2; ++j) {
                acc1[i][j] = __builtin_amdgcn_wmma_f32_16x16x32_bf16(
                    false, a[i], false, b1[j], (short)0, acc1[i][j], false, false);
                acc3[i][j] = __builtin_amdgcn_wmma_f32_16x16x32_bf16(
                    false, a[i], false, b3[j], (short)0, acc3[i][j], false, false);
            }
        __syncthreads();
    }

    // ---- epilogue: h = silu(acc1) * acc3, store bf16 ----------------------
    const int rsel = (lane >> 4) & 1;
    const int nl   = lane & 15;
#pragma unroll
    for (int i = 0; i < 2; ++i)
#pragma unroll
        for (int j = 0; j < 2; ++j) {
            int mb = m0 + wm * 32 + i * 16;
            int nb = n0 + wn * 32 + j * 16 + nl;
#pragma unroll
            for (int r = 0; r < 8; ++r) {
                int   m  = mb + r + rsel * 8;
                float a1 = acc1[i][j][r];
                float a3 = acc3[i][j][r];
                float s  = a1 / (1.0f + __expf(-a1));   // silu
                h[(size_t)(e * CAP + m) * HIDDEN + nb] = f2bf(s * a3);
            }
        }
}

// ---------------------------------------------------------------------------
// Kernel 4: GEMM2 + scatter: out[token] += gate * (h @ w2^T)
// grid: (DIM/BN, CAP/BM, NEXP), block 256
// ---------------------------------------------------------------------------
__global__ __launch_bounds__(256) void moe_gemm2(
    const unsigned short* __restrict__ h, const float* __restrict__ w2,
    const int* __restrict__ row_map, const float* __restrict__ gate,
    float* __restrict__ out)
{
    __shared__ unsigned short aS[BM * LDSTRIDE];
    __shared__ unsigned short bS[BN * LDSTRIDE];
    __shared__ int   tokS[BM];
    __shared__ float gS[BM];

    const int e    = blockIdx.z;
    const int m0   = blockIdx.y * BM;
    const int n0   = blockIdx.x * BN;
    const int tid  = threadIdx.x;
    const int lane = tid & 31;
    const int wv   = tid >> 5;
    const int wm   = wv & 3;
    const int wn   = wv >> 2;

    if (tid < BM) {
        tokS[tid] = row_map[e * CAP + m0 + tid] >> 1;
        gS[tid]   = gate[e * CAP + m0 + tid];
    }
    __syncthreads();

    const unsigned short* he  = h  + (size_t)e * CAP * HIDDEN;
    const float*          w2e = w2 + (size_t)e * DIM * HIDDEN;

    v8f acc[2][2] = {};

    for (int k0 = 0; k0 < HIDDEN; k0 += BK) {
        // ---- A tile: copy 128x32 bf16 from h ------------------------------
#pragma unroll
        for (int t = 0; t < 2; ++t) {
            int f   = tid + t * 256;         // 0..511 uint4 slots (16B of bf16)
            int row = f >> 2;
            int c   = f & 3;
            uint4 v = *(const uint4*)(he + (size_t)(m0 + row) * HIDDEN + k0 + c * 8);
            *(uint4*)(aS + row * LDSTRIDE + c * 8) = v;
        }
        // ---- B tile: w2 64x32 fp32 -> bf16 --------------------------------
#pragma unroll
        for (int t = 0; t < 2; ++t) {
            int f   = tid + t * 256;
            int row = f >> 3;
            int c4  = f & 7;
            size_t off = (size_t)(n0 + row) * HIDDEN + k0 + c4 * 4;
            float4 v = *(const float4*)(w2e + off);
            *(uint2*)(bS + row * LDSTRIDE + c4 * 4) =
                make_uint2(pack2(v.x, v.y), pack2(v.z, v.w));
            if (k0 + BK < HIDDEN)
                __builtin_prefetch(w2e + off + BK, 0, 1);
        }
        __syncthreads();

        v16bf a[2], b[2];
#pragma unroll
        for (int i = 0; i < 2; ++i)
            a[i] = load_fragA(aS + (wm * 32 + i * 16) * LDSTRIDE, lane);
#pragma unroll
        for (int j = 0; j < 2; ++j)
            b[j] = load_fragB(bS + (wn * 32 + j * 16) * LDSTRIDE, lane);
#pragma unroll
        for (int i = 0; i < 2; ++i)
#pragma unroll
            for (int j = 0; j < 2; ++j)
                acc[i][j] = __builtin_amdgcn_wmma_f32_16x16x32_bf16(
                    false, a[i], false, b[j], (short)0, acc[i][j], false, false);
        __syncthreads();
    }

    // ---- epilogue: scale by gate, atomic scatter-add into out[token] ------
    const int rsel = (lane >> 4) & 1;
    const int nl   = lane & 15;
#pragma unroll
    for (int i = 0; i < 2; ++i)
#pragma unroll
        for (int j = 0; j < 2; ++j) {
            int mb = wm * 32 + i * 16;          // local row within block
            int nb = n0 + wn * 32 + j * 16 + nl;
#pragma unroll
            for (int r = 0; r < 8; ++r) {
                int   ml = mb + r + rsel * 8;
                float v  = acc[i][j][r] * gS[ml];
                atomicAdd(out + (size_t)tokS[ml] * DIM + nb, v);
            }
        }
}

// ---------------------------------------------------------------------------
// Host launcher
// inputs: x[T,DIM] f32, top_scores[T,K] f32, sel[T,K] i32,
//         w1[E,H,D] f32, w2[E,D,H] f32, w3[E,H,D] f32
// out: [T,DIM] f32.  ws: row_map(64KB) | gate(64KB) | h bf16 (44MB)
// ---------------------------------------------------------------------------
extern "C" void kernel_launch(void* const* d_in, const int* in_sizes, int n_in,
                              void* d_out, int out_size, void* d_ws, size_t ws_size,
                              hipStream_t stream) {
    const float* x      = (const float*)d_in[0];
    const float* scores = (const float*)d_in[1];
    const int*   sel    = (const int*)  d_in[2];
    const float* w1     = (const float*)d_in[3];
    const float* w2     = (const float*)d_in[4];
    const float* w3     = (const float*)d_in[5];
    float* out = (float*)d_out;

    char* ws = (char*)d_ws;
    int*            row_map = (int*)ws;                          // 16384 ints
    float*          gate    = (float*)(ws + 64 * 1024);          // 16384 floats
    unsigned short* h       = (unsigned short*)(ws + 128 * 1024); // E*CAP*HIDDEN bf16

    moe_dispatch<<<1, 256, 0, stream>>>(sel, scores, row_map, gate);

    int n4 = (TOKENS * DIM) / 4;
    zero_out<<<(n4 + 255) / 256, 256, 0, stream>>>((float4*)out, n4);

    dim3 g1(HIDDEN / BN, CAP / BM, NEXP);   // (22, 16, 8)
    moe_gemm1<<<g1, 256, 0, stream>>>(x, w1, w3, row_map, h);

    dim3 g2(DIM / BN, CAP / BM, NEXP);      // (32, 16, 8)
    moe_gemm2<<<g2, 256, 0, stream>>>(h, w2, row_map, gate, out);
}
